// NonLocalBlock_50397146251812
// MI455X (gfx1250) — matrix operand order repaired and checked by reading, not compiled, not run on hardware
//
#include <hip/hip_runtime.h>
#include <hip/hip_bf16.h>

// ---------------------------------------------------------------------------
// CDNA5 (gfx1250) NonLocal block: WMMA f32_16x16x32_f16 everywhere,
// plus async global->LDS staging (ASYNCcnt) for the attention K tiles.
// ---------------------------------------------------------------------------

typedef __attribute__((ext_vector_type(16))) _Float16 v16h;
typedef __attribute__((ext_vector_type(8)))  _Float16 v8h;
typedef __attribute__((ext_vector_type(8)))  float    v8f;

#define BATCH 8
#define CCH   256
#define NPOS  4096   // 64*64
#define MPOS  1024   // 32*32
#define ICH   64

__device__ __forceinline__ v8f wmma_f16(v16h a, v16h b, v8f c) {
    // 8 args: (neg_a, A, neg_b, B, c_mod, C, reuse_a, reuse_b)
    return __builtin_amdgcn_wmma_f32_16x16x32_f16(false, a, false, b,
                                                  (short)0, c, false, false);
}

// A fragment: 16x32 (MxK), memory row-major rows=M, K contiguous, 16B aligned.
// Lane r(=lane&15) holds row M=r; half(=lane>>4) selects K {0..7,16..23} vs {8..15,24..31}.
__device__ __forceinline__ v16h load_a_frag(const _Float16* base, int ld, int lane) {
    const int r = lane & 15, h = lane >> 4;
    const _Float16* p = base + (size_t)r * ld + h * 8;
    v8h lo = *reinterpret_cast<const v8h*>(p);
    v8h hi = *reinterpret_cast<const v8h*>(p + 16);
    v16h a;
#pragma unroll
    for (int e = 0; e < 8; ++e) { a[e] = lo[e]; a[8 + e] = hi[e]; }
    return a;
}

// B fragment: 32x16 (KxN), memory stored N-major (row = N column index, K contiguous).
// Lane r holds column N=r; half selects K {0..7,16..23} vs {8..15,24..31}.
__device__ __forceinline__ v16h load_b_frag(const _Float16* base, int ld, int lane) {
    const int r = lane & 15, h = lane >> 4;
    const _Float16* p = base + (size_t)r * ld + h * 8;
    v8h lo = *reinterpret_cast<const v8h*>(p);
    v8h hi = *reinterpret_cast<const v8h*>(p + 16);
    v16h b;
#pragma unroll
    for (int e = 0; e < 8; ++e) { b[e] = lo[e]; b[8 + e] = hi[e]; }
    return b;
}

__device__ __forceinline__ v8f zero8() {
    v8f z = {0.f, 0.f, 0.f, 0.f, 0.f, 0.f, 0.f, 0.f};
    return z;
}

// Async copy of one 2048-byte K tile (16 rows x 64 f16, contiguous) into LDS.
// Each GLOBAL_LOAD_ASYNC_TO_LDS_B128 moves 16B per lane (512B/wave); 4 issues.
// lds_off is the wave-relative LDS byte offset (low 32 bits of the flat ptr).
__device__ __forceinline__ void async_copy_2k(const _Float16* gsrc,
                                              unsigned lds_off, int lane) {
    unsigned long long g = (unsigned long long)gsrc + (unsigned)(lane * 16);
    unsigned l = lds_off + (unsigned)(lane * 16);
#pragma unroll
    for (int j = 0; j < 4; ++j) {
        asm volatile("global_load_async_to_lds_b128 %0, %1, off"
                     :
                     : "v"(l + j * 512), "v"(g + (unsigned long long)(j * 512))
                     : "memory");
    }
}

// ---------------------------------------------------------------------------
// Kernel 1a: weights f32 -> f16
// ---------------------------------------------------------------------------
__global__ void pack_weights_kernel(const float* __restrict__ gw,
                                    const float* __restrict__ tw,
                                    const float* __restrict__ pw,
                                    const float* __restrict__ Ww,
                                    _Float16* __restrict__ wg,
                                    _Float16* __restrict__ wt,
                                    _Float16* __restrict__ wp,
                                    _Float16* __restrict__ ww) {
    int i = blockIdx.x * blockDim.x + threadIdx.x;
    if (i < ICH * CCH) {
        wg[i] = (_Float16)gw[i];
        wt[i] = (_Float16)tw[i];
        wp[i] = (_Float16)pw[i];
        ww[i] = (_Float16)Ww[i];   // W_w is [256][64] = same element count
    }
}

// ---------------------------------------------------------------------------
// Kernel 1b: x [b,c,p] f32 -> xt [b,p,c] f16 (LDS-tiled transpose)
// grid (128, 8, B), block (32, 8)
// ---------------------------------------------------------------------------
__global__ void transpose_x_kernel(const float* __restrict__ x,
                                   _Float16* __restrict__ xt) {
    __shared__ float tile[32][33];
    const int b  = blockIdx.z;
    const int p0 = blockIdx.x * 32;
    const int c0 = blockIdx.y * 32;
    const int tx = threadIdx.x, ty = threadIdx.y;
#pragma unroll
    for (int j = 0; j < 4; ++j) {
        int c = c0 + ty + j * 8;
        tile[ty + j * 8][tx] = x[((size_t)b * CCH + c) * NPOS + p0 + tx];
    }
    __syncthreads();
#pragma unroll
    for (int j = 0; j < 4; ++j) {
        int p = p0 + ty + j * 8;
        xt[((size_t)b * NPOS + p) * CCH + c0 + tx] = (_Float16)tile[tx][ty + j * 8];
    }
}

// ---------------------------------------------------------------------------
// Kernel 2: fused theta/g/phi 1x1 conv. One wave per 16(pos) x 16(ch) tile.
// Out[p, ic] = sum_c xt[p,c] * w[ic,c] + bias[ic]
// grid (B*256, 12): y = weight-set(3) * ic-tile(4)
// ---------------------------------------------------------------------------
__global__ void conv_qkv_kernel(const _Float16* __restrict__ xt,
                                const _Float16* __restrict__ wq,
                                const _Float16* __restrict__ wg,
                                const _Float16* __restrict__ wp,
                                const float* __restrict__ bq,
                                const float* __restrict__ bg,
                                const float* __restrict__ bp,
                                _Float16* __restrict__ qo,
                                _Float16* __restrict__ gf,
                                _Float16* __restrict__ pf) {
    const int lane = threadIdx.x & 31;
    const int r = lane & 15, h = lane >> 4;
    const int b  = blockIdx.x >> 8;
    const int p0 = (blockIdx.x & 255) * 16;
    const int nt = blockIdx.y;
    const int set = nt >> 2;
    const int ic0 = (nt & 3) * 16;

    const _Float16* w    = (set == 0) ? wq : (set == 1) ? wg : wp;
    const float*    bias = (set == 0) ? bq : (set == 1) ? bg : bp;
    _Float16*       out  = (set == 0) ? qo : (set == 1) ? gf : pf;

    const _Float16* abase = xt + ((size_t)b * NPOS + p0) * CCH;
    const _Float16* bbase = w + (size_t)ic0 * CCH;

    v8f acc = zero8();
#pragma unroll
    for (int ks = 0; ks < 8; ++ks) {      // K = 256 = 8 * 32
        __builtin_prefetch(abase + (ks + 1) * 32, 0, 1);
        v16h a  = load_a_frag(abase + ks * 32, CCH, lane);
        v16h bb = load_b_frag(bbase + ks * 32, CCH, lane);
        acc = wmma_f16(a, bb, acc);
    }
    const float bv = bias[ic0 + r];
#pragma unroll
    for (int i = 0; i < 8; ++i) {
        out[((size_t)b * NPOS + p0 + h * 8 + i) * ICH + ic0 + r] =
            (_Float16)(acc[i] + bv);
    }
}

// ---------------------------------------------------------------------------
// Kernel 3: 2x2 maxpool. gf/pf are [b, p(4096), ic] f16.
//  -> vt [b, ic, m]  (B-frag layout for P*V)
//  -> kb [b, m, ic]  (B-frag layout for Q*K^T)
// ---------------------------------------------------------------------------
__global__ void pool_kernel(const _Float16* __restrict__ gf,
                            const _Float16* __restrict__ pf,
                            _Float16* __restrict__ kb,
                            _Float16* __restrict__ vt) {
    int idx = blockIdx.x * blockDim.x + threadIdx.x;
    if (idx >= BATCH * MPOS * ICH) return;
    const int ic = idx & 63;
    const int m  = (idx >> 6) & 1023;
    const int b  = idx >> 16;
    const int my = m >> 5, mx = m & 31;
    const int p00 = (my * 2) * 64 + mx * 2;
    const size_t base = (size_t)b * NPOS * ICH;

    float g0 = (float)gf[base + (size_t)(p00     ) * ICH + ic];
    float g1 = (float)gf[base + (size_t)(p00 + 1 ) * ICH + ic];
    float g2 = (float)gf[base + (size_t)(p00 + 64) * ICH + ic];
    float g3 = (float)gf[base + (size_t)(p00 + 65) * ICH + ic];
    float q0 = (float)pf[base + (size_t)(p00     ) * ICH + ic];
    float q1 = (float)pf[base + (size_t)(p00 + 1 ) * ICH + ic];
    float q2 = (float)pf[base + (size_t)(p00 + 64) * ICH + ic];
    float q3 = (float)pf[base + (size_t)(p00 + 65) * ICH + ic];

    vt[((size_t)b * ICH + ic) * MPOS + m] =
        (_Float16)fmaxf(fmaxf(g0, g1), fmaxf(g2, g3));
    kb[((size_t)b * MPOS + m) * ICH + ic] =
        (_Float16)fmaxf(fmaxf(q0, q1), fmaxf(q2, q3));
}

// ---------------------------------------------------------------------------
// Kernel 4: attention. One wave handles a 16-query tile.
// S[16][1024] f32 in LDS, softmax, P f16 in LDS, P*V via WMMA.
// K tiles are double-buffered into LDS with GLOBAL_LOAD_ASYNC_TO_LDS_B128.
// grid (B*256), block 32.
// LDS: 64KB S + 32KB P + 64B sums + 4KB K staging = 102464 bytes.
// ---------------------------------------------------------------------------
#define ATTN_S_BYTES   (16 * 1024 * 4)
#define ATTN_P_BYTES   (16 * 1024 * 2)
#define ATTN_SUM_BYTES 64
#define ATTN_KST_OFF   (ATTN_S_BYTES + ATTN_P_BYTES + ATTN_SUM_BYTES)
#define ATTN_LDS_BYTES (ATTN_KST_OFF + 2 * 2048)

__global__ void attn_kernel(const _Float16* __restrict__ q,
                            const _Float16* __restrict__ kb,
                            const _Float16* __restrict__ vt,
                            _Float16* __restrict__ att) {
    extern __shared__ char smem[];
    float*    S    = reinterpret_cast<float*>(smem);                         // [16][1024]
    _Float16* P    = reinterpret_cast<_Float16*>(smem + ATTN_S_BYTES);       // [16][1024]
    float*    ssum = reinterpret_cast<float*>(smem + ATTN_S_BYTES + ATTN_P_BYTES);
    _Float16* kst  = reinterpret_cast<_Float16*>(smem + ATTN_KST_OFF);       // 2 x 1024 f16
    const unsigned kst_off = (unsigned)(unsigned long long)kst;  // wave-rel LDS byte addr

    const int lane = threadIdx.x & 31;
    const int r = lane & 15, h = lane >> 4;
    const int b  = blockIdx.x >> 8;
    const int p0 = (blockIdx.x & 255) * 16;

    // Q tile A-fragments (16 x 64 = two K-steps), row-major [p, ic]
    const _Float16* qb = q + ((size_t)b * NPOS + p0) * ICH;
    const v16h aq0 = load_a_frag(qb,      ICH, lane);
    const v16h aq1 = load_a_frag(qb + 32, ICH, lane);

    // S = Q K^T : 64 key-tiles of 16, K = ic(64) = 2 WMMA steps each.
    // Each 16x64 K tile is 2048 contiguous bytes in kb; stage via async DMA.
    const _Float16* kbase = kb + (size_t)b * MPOS * ICH;
    async_copy_2k(kbase, kst_off, lane);                       // tile 0 -> buf 0
    for (int nt = 0; nt < 64; ++nt) {
        if (nt + 1 < 64) {
            async_copy_2k(kbase + (size_t)(nt + 1) * 16 * ICH,
                          kst_off + (unsigned)(((nt + 1) & 1) * 2048), lane);
            asm volatile("s_wait_asynccnt 0x4" ::: "memory");  // tile nt landed
        } else {
            asm volatile("s_wait_asynccnt 0x0" ::: "memory");
        }
        const _Float16* kt = kst + (size_t)(nt & 1) * 1024;
        v8f acc = zero8();
        v16h b0 = load_b_frag(kt,      ICH, lane);
        v16h b1 = load_b_frag(kt + 32, ICH, lane);
        acc = wmma_f16(aq0, b0, acc);
        acc = wmma_f16(aq1, b1, acc);
#pragma unroll
        for (int i = 0; i < 8; ++i)
            S[(i + h * 8) * 1024 + nt * 16 + r] = acc[i];
    }
    __syncthreads();

    // Row softmax: lane handles row r, half of the 1024 columns.
    {
        const int c0 = h * 512;
        float mx = -3.0e38f;
        for (int c = 0; c < 512; ++c)
            mx = fmaxf(mx, S[r * 1024 + c0 + c]);
        mx = fmaxf(mx, __shfl_xor(mx, 16, 32));
        float sum = 0.f;
        for (int c = 0; c < 512; ++c) {
            float e = __expf(S[r * 1024 + c0 + c] - mx);
            P[r * 1024 + c0 + c] = (_Float16)e;
            sum += e;
        }
        sum += __shfl_xor(sum, 16, 32);
        if (h == 0) ssum[r] = sum;
    }
    __syncthreads();

    // attended = P * V : 4 ic-tiles, K = 1024 = 32 WMMA steps
    const _Float16* vbase = vt + (size_t)b * ICH * MPOS;
    v8f o0 = zero8(), o1 = zero8(), o2 = zero8(), o3 = zero8();
    for (int kc = 0; kc < 32; ++kc) {
        v16h ap = load_a_frag(P + kc * 32, 1024, lane);
        o0 = wmma_f16(ap, load_b_frag(vbase + (size_t)( 0) * MPOS + kc * 32, MPOS, lane), o0);
        o1 = wmma_f16(ap, load_b_frag(vbase + (size_t)(16) * MPOS + kc * 32, MPOS, lane), o1);
        o2 = wmma_f16(ap, load_b_frag(vbase + (size_t)(32) * MPOS + kc * 32, MPOS, lane), o2);
        o3 = wmma_f16(ap, load_b_frag(vbase + (size_t)(48) * MPOS + kc * 32, MPOS, lane), o3);
    }
    // Normalize by row sums and store attended [b, p, ic] f16.
#pragma unroll
    for (int i = 0; i < 8; ++i) {
        const float inv = 1.0f / ssum[i + h * 8];
        const size_t row = ((size_t)b * NPOS + p0 + h * 8 + i) * ICH;
        att[row +  0 + r] = (_Float16)(o0[i] * inv);
        att[row + 16 + r] = (_Float16)(o1[i] * inv);
        att[row + 32 + r] = (_Float16)(o2[i] * inv);
        att[row + 48 + r] = (_Float16)(o3[i] * inv);
    }
}

// ---------------------------------------------------------------------------
// Kernel 5: out[b,c,p] = sum_ic att[p,ic]*W_w[c,ic] + W_b[c] + x[b,c,p]
// D rows = p -> contiguous f32 stores per output channel. grid (B*256, 16)
// ---------------------------------------------------------------------------
__global__ void conv_out_kernel(const _Float16* __restrict__ att,
                                const _Float16* __restrict__ ww,
                                const float* __restrict__ wb,
                                const float* __restrict__ xin,
                                float* __restrict__ out) {
    const int lane = threadIdx.x & 31;
    const int r = lane & 15, h = lane >> 4;
    const int b  = blockIdx.x >> 8;
    const int p0 = (blockIdx.x & 255) * 16;
    const int c0 = blockIdx.y * 16;

    const _Float16* abase = att + ((size_t)b * NPOS + p0) * ICH;
    const _Float16* bbase = ww + (size_t)c0 * ICH;

    v8f acc = zero8();
    acc = wmma_f16(load_a_frag(abase,      ICH, lane),
                   load_b_frag(bbase,      ICH, lane), acc);
    acc = wmma_f16(load_a_frag(abase + 32, ICH, lane),
                   load_b_frag(bbase + 32, ICH, lane), acc);

    const int c = c0 + r;
    const float bias = wb[c];
    const size_t rowbase = ((size_t)b * CCH + c) * NPOS + p0 + h * 8;
#pragma unroll
    for (int i = 0; i < 8; ++i)
        out[rowbase + i] = acc[i] + bias + xin[rowbase + i];
}

// ---------------------------------------------------------------------------
// Launch
// ---------------------------------------------------------------------------
extern "C" void kernel_launch(void* const* d_in, const int* in_sizes, int n_in,
                              void* d_out, int out_size, void* d_ws, size_t ws_size,
                              hipStream_t stream) {
    (void)in_sizes; (void)n_in; (void)out_size; (void)ws_size;

    const float* x   = (const float*)d_in[0];
    const float* g_w = (const float*)d_in[1];
    const float* g_b = (const float*)d_in[2];
    const float* t_w = (const float*)d_in[3];
    const float* t_b = (const float*)d_in[4];
    const float* p_w = (const float*)d_in[5];
    const float* p_b = (const float*)d_in[6];
    const float* W_w = (const float*)d_in[7];
    const float* W_b = (const float*)d_in[8];
    float* out = (float*)d_out;

    char* ws = (char*)d_ws;
    size_t off = 0;
    auto alloc = [&](size_t bytes) {
        char* p = ws + off;
        off += (bytes + 255) & ~(size_t)255;
        return p;
    };
    _Float16* xt  = (_Float16*)alloc((size_t)BATCH * NPOS * CCH * 2);  // 16 MiB
    _Float16* wgp = (_Float16*)alloc((size_t)ICH * CCH * 2);
    _Float16* wtp = (_Float16*)alloc((size_t)ICH * CCH * 2);
    _Float16* wpp = (_Float16*)alloc((size_t)ICH * CCH * 2);
    _Float16* wwp = (_Float16*)alloc((size_t)CCH * ICH * 2);
    _Float16* q   = (_Float16*)alloc((size_t)BATCH * NPOS * ICH * 2);  // 4 MiB
    _Float16* gf  = (_Float16*)alloc((size_t)BATCH * NPOS * ICH * 2);
    _Float16* pf  = (_Float16*)alloc((size_t)BATCH * NPOS * ICH * 2);
    _Float16* kbf = (_Float16*)alloc((size_t)BATCH * MPOS * ICH * 2);  // 1 MiB
    _Float16* vtf = (_Float16*)alloc((size_t)BATCH * ICH * MPOS * 2);
    _Float16* att = (_Float16*)alloc((size_t)BATCH * NPOS * ICH * 2);

    // 1) pack weights + transpose x to f16
    pack_weights_kernel<<<(ICH * CCH + 255) / 256, 256, 0, stream>>>(
        g_w, t_w, p_w, W_w, wgp, wtp, wpp, wwp);
    transpose_x_kernel<<<dim3(NPOS / 32, CCH / 32, BATCH), dim3(32, 8), 0, stream>>>(x, xt);

    // 2) fused theta/g/phi 1x1 convs (WMMA GEMM)
    conv_qkv_kernel<<<dim3(BATCH * (NPOS / 16), 12), 32, 0, stream>>>(
        xt, wtp, wgp, wpp, t_b, g_b, p_b, q, gf, pf);

    // 3) 2x2 maxpool into WMMA-friendly K / V layouts
    pool_kernel<<<(BATCH * MPOS * ICH + 255) / 256, 256, 0, stream>>>(gf, pf, kbf, vtf);

    // 4) attention (softmax(QK^T) V), one wave per 16-query tile
    attn_kernel<<<BATCH * (NPOS / 16), 32, ATTN_LDS_BYTES, stream>>>(q, kbf, vtf, att);

    // 5) output 1x1 conv + bias + residual
    conv_out_kernel<<<dim3(BATCH * (NPOS / 16), CCH / 16), 32, 0, stream>>>(
        att, wwp, W_b, x, out);
}
